// CrossViewEnhancement_66803921322384
// MI455X (gfx1250) — compile-verified, matrix operand order checked by reading, not run on hardware
//
#include <hip/hip_runtime.h>
#include <hip/hip_bf16.h>

// ---------------------------------------------------------------------------
// CrossViewEnhancement for MI455X (gfx1250, wave32, WMMA, async-LDS copies).
//
// Math restructuring:
//   S[b,c,x]   = sum_y front_x[b,c,y,x]; row0/rowH captured alongside.
//   k[b,c8,w2] = kb + sum_c kw * (S[2w2]+S[2w2+1]) / 256
//   vsum[b,c,w2] = 64*vb + 1/4 * sum_{ci,ky,kx} vw * (Z_ky[2w2+kx-1]+Z_ky[2w2+kx])
//                  with Z_0 = S-rowH, Z_1 = S, Z_2 = S-row0   (y-conv collapsed)
//   e[b,h2,w2] = kqb + sum_c m[b,c,w2]*bevpool ; m = k^T qw ; then L2-norm over h.
//   cat (NHWC bf16 [B,130,130,512], zero halo): 240 bev ch + 256 T ch + 16 pad.
//   out[:,16:] = conv3x3(cat, fw) + fb via implicit GEMM,
//                v_wmma_f32_16x16x32_bf16 (f32 accum), A/B staged to LDS with
//                global_load_async_to_lds_b128 (ASYNCcnt, 4-buffer pipeline).
//   out[:,:16] = bev_x[:,:16].
// ---------------------------------------------------------------------------

typedef __bf16 bf16;
typedef __attribute__((ext_vector_type(16))) __bf16 v16bf;
typedef __attribute__((ext_vector_type(8)))  float  v8f;

static constexpr int B_   = 8;
static constexpr int C_   = 256;
static constexpr int H_   = 128;
static constexpr int W_   = 128;
static constexpr int C8_  = 32;
static constexpr int HP   = 64;     // pooled
static constexpr int WP   = 64;
static constexpr int CIN  = 512;    // padded NHWC channels for final conv
static constexpr int NOUT = 240;    // final conv output channels
static constexpr int H2   = 130;    // padded spatial (halo for 3x3)
static constexpr int W2   = 130;
static constexpr int KSTEPS = (9 * CIN) / 32;  // 144 K-steps of 32
static constexpr int NTILES = 16;              // 15 real N-tiles + 1 zero pad

__device__ __forceinline__ unsigned lds_off(const void* p) {
  // On AMDGPU a generic pointer to LDS carries the LDS byte offset in its
  // low 32 bits (high bits = shared aperture).
  return (unsigned)(unsigned long long)(p);
}

// ---------------------------------------------------------------------------
// 1. Column reduction over front_x:  S, row0, rowH   [B,C,W]
// ---------------------------------------------------------------------------
__global__ void reduce_front_k(const float* __restrict__ front,
                               float* __restrict__ S,
                               float* __restrict__ R0,
                               float* __restrict__ RH) {
  int bc = blockIdx.x;            // b*C + c
  int x  = threadIdx.x;           // 0..127
  const float* col = front + (size_t)bc * H_ * W_ + x;
  float s = 0.f, r0 = 0.f, rh = 0.f;
  for (int y = 0; y < H_; ++y) {
    float v = col[(size_t)y * W_];
    s += v;
    if (y == 0)      r0 = v;
    if (y == H_ - 1) rh = v;
  }
  size_t o = (size_t)bc * W_ + x;
  S[o] = s; R0[o] = r0; RH[o] = rh;
}

// ---------------------------------------------------------------------------
// 2. k[b,c8,w2] from S
// ---------------------------------------------------------------------------
__global__ void compute_k_k(const float* __restrict__ S,
                            const float* __restrict__ kw,
                            const float* __restrict__ kb,
                            float* __restrict__ kbuf) {
  __shared__ float s[C_];
  int b = blockIdx.x >> 6, w2 = blockIdx.x & 63;
  int c = threadIdx.x;
  const float* Sp = S + ((size_t)(b * C_ + c)) * W_ + 2 * w2;
  s[c] = (Sp[0] + Sp[1]) * (1.f / 256.f);
  __syncthreads();
  if (c < C8_) {
    float acc = kb[c];
    for (int i = 0; i < C_; ++i) acc += kw[c * C_ + i] * s[i];
    kbuf[((size_t)b * C8_ + c) * WP + w2] = acc;
  }
}

// ---------------------------------------------------------------------------
// 3. vsum[b,co,w2] — collapsed-y 3x3 conv on (S, row0, rowH)
// ---------------------------------------------------------------------------
__global__ void compute_vsum_k(const float* __restrict__ S,
                               const float* __restrict__ R0,
                               const float* __restrict__ RH,
                               const float* __restrict__ vw,
                               const float* __restrict__ vb,
                               float* __restrict__ vsum) {
  __shared__ float pp[4][C_][9];          // 36 KB
  int b = blockIdx.x >> 4, grp = blockIdx.x & 15;   // group of 4 w2
  int ci = threadIdx.x;
  const float* Sr  = S  + ((size_t)(b * C_ + ci)) * W_;
  const float* Tr0 = R0 + ((size_t)(b * C_ + ci)) * W_;
  const float* TrH = RH + ((size_t)(b * C_ + ci)) * W_;
  for (int i = 0; i < 4; ++i) {
    int w2 = grp * 4 + i;
    float z[3][4];
    for (int d = 0; d < 4; ++d) {
      int col = 2 * w2 - 1 + d;
      float sv = 0.f, r0 = 0.f, rh = 0.f;
      if ((unsigned)col < (unsigned)W_) { sv = Sr[col]; r0 = Tr0[col]; rh = TrH[col]; }
      z[0][d] = sv - rh;   // ky = 0 (offset -1)
      z[1][d] = sv;        // ky = 1
      z[2][d] = sv - r0;   // ky = 2 (offset +1)
    }
    for (int ky = 0; ky < 3; ++ky)
      for (int kx = 0; kx < 3; ++kx)
        pp[i][ci][ky * 3 + kx] = 0.25f * (z[ky][kx] + z[ky][kx + 1]);
  }
  __syncthreads();
  int co = threadIdx.x;
  float acc[4];
  float bias = 64.f * vb[co];
  for (int i = 0; i < 4; ++i) acc[i] = bias;
  const float* wp = vw + (size_t)co * C_ * 9;
  for (int cc = 0; cc < C_; ++cc) {
    float w9[9];
    #pragma unroll
    for (int j = 0; j < 9; ++j) w9[j] = wp[cc * 9 + j];
    #pragma unroll
    for (int i = 0; i < 4; ++i) {
      float a = 0.f;
      #pragma unroll
      for (int j = 0; j < 9; ++j) a += w9[j] * pp[i][cc][j];
      acc[i] += a;
    }
  }
  for (int i = 0; i < 4; ++i)
    vsum[((size_t)b * C_ + co) * WP + grp * 4 + i] = acc[i];
}

// ---------------------------------------------------------------------------
// 4. m[b,c,w2] = sum_c8 k*qw ;  kqb[b,w2] = sum_c8 k*qb
// ---------------------------------------------------------------------------
__global__ void compute_m_k(const float* __restrict__ kbuf,
                            const float* __restrict__ qw,
                            const float* __restrict__ qb,
                            float* __restrict__ mbuf,
                            float* __restrict__ kqb) {
  __shared__ float kk[C8_];
  int b = blockIdx.x >> 6, w2 = blockIdx.x & 63;
  int t = threadIdx.x;
  if (t < C8_) kk[t] = kbuf[((size_t)b * C8_ + t) * WP + w2];
  __syncthreads();
  float acc = 0.f;
  for (int c8 = 0; c8 < C8_; ++c8) acc += kk[c8] * qw[c8 * C_ + t];
  mbuf[((size_t)b * C_ + t) * WP + w2] = acc;
  if (t == 0) {
    float a = 0.f;
    for (int c8 = 0; c8 < C8_; ++c8) a += kk[c8] * qb[c8];
    kqb[b * WP + w2] = a;
  }
}

// ---------------------------------------------------------------------------
// 5. e_raw[b,h2,w2] = kqb + sum_c m * bevpool   (streams bev_x once)
// ---------------------------------------------------------------------------
__global__ void compute_e_k(const float* __restrict__ bev,
                            const float* __restrict__ mbuf,
                            const float* __restrict__ kqb,
                            float* __restrict__ ebuf) {
  __shared__ float red[4][WP];
  int b = blockIdx.x >> 6, h2 = blockIdx.x & 63;
  int w2 = threadIdx.x & 63, cg = threadIdx.x >> 6;   // 4 channel groups of 64
  float acc = 0.f;
  for (int c = cg * 64; c < cg * 64 + 64; ++c) {
    const float* p = bev + (((size_t)(b * C_ + c)) * H_ + 2 * h2) * W_ + 2 * w2;
    float v = (p[0] + p[1] + p[W_] + p[W_ + 1]) * 0.25f;
    acc += mbuf[((size_t)b * C_ + c) * WP + w2] * v;
  }
  red[cg][w2] = acc;
  __syncthreads();
  if (threadIdx.x < WP) {
    int w = threadIdx.x;
    ebuf[((size_t)b * HP + h2) * WP + w] =
        kqb[b * WP + w] + red[0][w] + red[1][w] + red[2][w] + red[3][w];
  }
}

// ---------------------------------------------------------------------------
// 6. L2-normalize e over h (per b, per column w2)
// ---------------------------------------------------------------------------
__global__ void normalize_e_k(float* __restrict__ ebuf) {
  __shared__ float sh[HP];
  int b = blockIdx.x >> 6, w2 = blockIdx.x & 63;
  int h = threadIdx.x;
  float v = ebuf[((size_t)b * HP + h) * WP + w2];
  sh[h] = v * v;
  __syncthreads();
  for (int s = 32; s > 0; s >>= 1) {
    if (h < s) sh[h] += sh[h + s];
    __syncthreads();
  }
  ebuf[((size_t)b * HP + h) * WP + w2] = v * rsqrtf(sh[0]);
}

// ---------------------------------------------------------------------------
// 7. Pack fw -> bf16 WMMA-B fragment order.
//    Bpack[kstep][ntile][lane][16]; lane<16: col=lane, K in {0..7,16..23};
//    lane>=16: col=lane-16, K in {8..15,24..31}  (ISA 7.12.2 16-bit layout).
//    K-index ordering: K = (ky*3+kx)*512 + c_padded (c>=496 zero pad).
// ---------------------------------------------------------------------------
__global__ void pack_weights_k(const float* __restrict__ fw, bf16* __restrict__ Bp) {
  int idx = blockIdx.x * blockDim.x + threadIdx.x;
  const int TOT = KSTEPS * NTILES * 32 * 16;
  if (idx >= TOT) return;
  int j     = idx & 15;
  int lane  = (idx >> 4) & 31;
  int ntile = (idx >> 9) & 15;
  int kstep = idx >> 13;
  int base  = (lane & 16) ? 8 : 0;
  int v = j >> 1, eo = j & 1;
  int kl = (v < 4) ? (base + 2 * v + eo) : (16 + base + 2 * (v - 4) + eo);
  int K   = kstep * 32 + kl;
  int c   = K & (CIN - 1);
  int kyx = K >> 9;                         // 0..8  -> (ky,kx)
  int n   = ntile * 16 + (lane & 15);
  float val = 0.f;
  if (n < NOUT && c < 496) val = fw[(size_t)n * 4464 + (size_t)c * 9 + kyx];
  Bp[idx] = (bf16)val;
}

// ---------------------------------------------------------------------------
// 8a. Zero the 1-pixel halo of cat [B,130,130,512]
// ---------------------------------------------------------------------------
__global__ void zero_halo_k(bf16* __restrict__ cat) {
  int cell = blockIdx.x;                 // b*516 + i
  int b = cell / 516, i = cell - b * 516;
  int yy, xx;
  if (i < 130)      { yy = 0;           xx = i; }
  else if (i < 260) { yy = H2 - 1;      xx = i - 130; }
  else if (i < 388) { yy = 1 + (i - 260); xx = 0; }
  else              { yy = 1 + (i - 388); xx = W2 - 1; }
  size_t basep = ((size_t)(b * H2 + yy) * W2 + xx) * CIN;
  uint2 z = {0u, 0u};
  *(uint2*)(cat + basep + threadIdx.x * 4) = z;   // 128 thr * 4 bf16 = 512
}

// ---------------------------------------------------------------------------
// 8b. Build cat interior: 240 bev channels, 256 T channels, 16 zero
// ---------------------------------------------------------------------------
__global__ void build_cat_k(const float* __restrict__ bev,
                            const float* __restrict__ ebuf,
                            const float* __restrict__ vsum,
                            bf16* __restrict__ cat) {
  __shared__ bf16  vsT[WP][C_];     // 32 KB transposed vsum[b]
  __shared__ float tile[32][W_];    // 16 KB NCHW->NHWC staging
  __shared__ float er[WP];
  int b = blockIdx.x >> 7, y = blockIdx.x & 127;
  int t = threadIdx.x;
  for (int i = 0; i < 64; ++i) {
    int idx = t + i * 256;          // 16384 = 256*64
    int ci = idx >> 6, w2 = idx & 63;
    vsT[w2][ci] = (bf16)vsum[((size_t)b * C_ + ci) * WP + w2];
  }
  if (t < WP) er[t] = ebuf[((size_t)b * HP + (y >> 1)) * WP + t];
  size_t rowbase = (size_t)(b * H2 + y + 1) * W2 + 1;   // padded origin
  // --- bev channels 16..255 -> cat channels 0..239 ---
  for (int cb = 0; cb < NOUT; cb += 32) {
    int csz = (NOUT - cb) < 32 ? (NOUT - cb) : 32;
    __syncthreads();
    for (int idx = t; idx < csz * W_; idx += 256) {
      int cl = idx >> 7, x = idx & 127;
      tile[cl][x] = bev[(((size_t)(b * C_ + 16 + cb + cl)) * H_ + y) * W_ + x];
    }
    __syncthreads();
    int x = t >> 1, half = t & 1;
    int c0 = half * 16;
    if (c0 < csz) {
      union { v16bf v; bf16 a[16]; } u;
      #pragma unroll
      for (int jj = 0; jj < 16; ++jj) u.a[jj] = (bf16)tile[c0 + jj][x];
      *(v16bf*)(cat + (rowbase + x) * CIN + cb + c0) = u.v;
    }
  }
  __syncthreads();
  // --- T channels: cat[240+ci] = e[b,y/2,x/2] * vsum[b,ci,x/2]; zero pad ---
  int x = t >> 1, half = t & 1;
  float ev = er[x >> 1];
  #pragma unroll
  for (int g = 0; g < 8; ++g) {
    int ci0 = half * 128 + g * 16;
    union { v16bf v; bf16 a[16]; } u;
    #pragma unroll
    for (int jj = 0; jj < 16; ++jj)
      u.a[jj] = (bf16)(ev * (float)vsT[x >> 1][ci0 + jj]);
    *(v16bf*)(cat + (rowbase + x) * CIN + NOUT + ci0) = u.v;
  }
  if (half == 0) {
    union { v16bf v; bf16 a[16]; } z;
    #pragma unroll
    for (int jj = 0; jj < 16; ++jj) z.a[jj] = (bf16)0.f;
    *(v16bf*)(cat + (rowbase + x) * CIN + 496) = z.v;
  }
}

// ---------------------------------------------------------------------------
// 9. out[:, :16] = bev[:, :16]
// ---------------------------------------------------------------------------
__global__ void copy16_k(const float4* __restrict__ bev4, float4* __restrict__ out4) {
  int idx = blockIdx.x * blockDim.x + threadIdx.x;
  if (idx >= B_ * 65536) return;
  int b = idx >> 16, i = idx & 65535;
  size_t off = (size_t)b * 1048576 + i;
  out4[off] = bev4[off];
}

// ---------------------------------------------------------------------------
// 10. Main implicit-GEMM conv3x3.
//     M=64 spatial x N=240 per WG; 8 waves = (M-half) x (4 N-tiles each).
//     Per K-step, A[64x32] and B-slab (16KB, fragment-packed) are staged to
//     LDS with global_load_async_to_lds_b128; 4 LDS buffers, issue depth 2,
//     one barrier per step; 8 x v_wmma_f32_16x16x32_bf16 per wave per step.
// ---------------------------------------------------------------------------
__global__ void __launch_bounds__(256, 2)
conv_main_k(const bf16* __restrict__ cat,
            const bf16* __restrict__ Bp,
            const float* __restrict__ fb,
            float* __restrict__ out) {
  __shared__ __align__(16) bf16 Alds[4][64 * 32];            //  4 x 4 KB
  __shared__ __align__(16) bf16 Blds[4][NTILES * 32 * 16];   //  4 x 16 KB
  int bid = blockIdx.x;
  int xh = bid & 1, y = (bid >> 1) & 127, b = bid >> 8;
  int t = threadIdx.x, lane = t & 31, wid = t >> 5;
  int mh = wid & 1, ng = wid >> 1;
  int am = t >> 2, achunk = t & 3;            // A staging role: row m, 8-elem chunk
  int k0 = (lane & 16) ? 8 : 0;               // 16-bit A/B per-lane K split

  unsigned aL[4], bL[4];
  #pragma unroll
  for (int i = 0; i < 4; ++i) {
    aL[i] = lds_off(&Alds[i][0]) + (unsigned)(am * 64 + achunk * 16);
    bL[i] = lds_off(&Blds[i][0]) + (unsigned)(t * 64);
  }
  unsigned long long catB = (unsigned long long)cat;
  unsigned long long bpB  = (unsigned long long)Bp;

  // spatial part of the per-thread A byte offset, per (ky,kx); 1 pixel = 1024 B
  unsigned arow = (unsigned)(b * H2 + y) * W2 + (unsigned)(xh * 64 + am);

  auto issue = [&](int kstep) {
    int buf = kstep & 3;
    int kyx = kstep >> 4;                 // uniform across the wave
    int cc  = kstep & 15;
    int ky = (kyx * 11) >> 5;             // kyx/3 for 0..8
    int kx = kyx - 3 * ky;
    unsigned aoff = (arow + (unsigned)(ky * W2 + kx)) * 1024u
                  + (unsigned)(cc * 64 + achunk * 16);
    asm volatile("global_load_async_to_lds_b128 %0, %1, %2"
                 :: "v"(aL[buf]), "v"(aoff), "s"(catB) : "memory");
    unsigned boff = (unsigned)kstep * 16384u + (unsigned)(t * 64);
    #pragma unroll
    for (int i = 0; i < 4; ++i)
      asm volatile("global_load_async_to_lds_b128 %0, %1, %2"
                   :: "v"(bL[buf] + i * 16), "v"(boff + i * 16), "s"(bpB)
                   : "memory");
  };

  v8f acc[2][4];
  #pragma unroll
  for (int s = 0; s < 2; ++s)
    #pragma unroll
    for (int j = 0; j < 4; ++j)
      acc[s][j] = (v8f){0.f, 0.f, 0.f, 0.f, 0.f, 0.f, 0.f, 0.f};

  issue(0);
  issue(1);
  for (int ks = 0; ks < KSTEPS; ++ks) {
    if (ks + 2 < KSTEPS) {
      issue(ks + 2);
      asm volatile("s_wait_asynccnt 0xa" ::: "memory");   // oldest 5 (step ks) done
    } else if (ks + 1 < KSTEPS) {
      asm volatile("s_wait_asynccnt 0x5" ::: "memory");
    } else {
      asm volatile("s_wait_asynccnt 0x0" ::: "memory");
    }
    __syncthreads();

    int cur = ks & 3;
    union U { v16bf v; uint4 u[2]; };
    U bfr[4];
    #pragma unroll
    for (int j = 0; j < 4; ++j) {
      const uint4* bp = (const uint4*)(&Blds[cur][((ng * 4 + j) * 32 + lane) * 16]);
      bfr[j].u[0] = bp[0];
      bfr[j].u[1] = bp[1];
    }
    #pragma unroll
    for (int s = 0; s < 2; ++s) {
      int mrow = mh * 32 + s * 16 + (lane & 15);
      const uint4* ap = (const uint4*)(&Alds[cur][mrow * 32 + k0]);
      U af;
      af.u[0] = ap[0];      // K = k0 .. k0+7
      af.u[1] = ap[2];      // K = k0+16 .. k0+23
      #pragma unroll
      for (int j = 0; j < 4; ++j)
        acc[s][j] = __builtin_amdgcn_wmma_f32_16x16x32_bf16(
            false, af.v, false, bfr[j].v, (short)0, acc[s][j], false, false);
    }
  }

  // ---- epilogue: bias + store (per-lane channel fixed -> 2x float4 along x) ----
  int nloc = lane & 15;
  int xlo = (lane & 16) ? 8 : 0;
  #pragma unroll
  for (int s = 0; s < 2; ++s) {
    #pragma unroll
    for (int j = 0; j < 4; ++j) {
      int ntile = ng * 4 + j;
      if (ntile >= 15) continue;          // pad tile
      int n = ntile * 16 + nloc;
      float bias = fb[n];
      int x = xh * 64 + mh * 32 + s * 16 + xlo;
      float* op = out + (((size_t)(b * C_ + 16 + n)) * H_ + y) * W_ + x;
      float4 lo = make_float4(acc[s][j][0] + bias, acc[s][j][1] + bias,
                              acc[s][j][2] + bias, acc[s][j][3] + bias);
      float4 hi = make_float4(acc[s][j][4] + bias, acc[s][j][5] + bias,
                              acc[s][j][6] + bias, acc[s][j][7] + bias);
      *(float4*)op = lo;
      *(float4*)(op + 4) = hi;
    }
  }
}

// ---------------------------------------------------------------------------
// Host launcher
// ---------------------------------------------------------------------------
extern "C" void kernel_launch(void* const* d_in, const int* in_sizes, int n_in,
                              void* d_out, int out_size, void* d_ws, size_t ws_size,
                              hipStream_t stream) {
  const float* front = (const float*)d_in[0];
  const float* bev   = (const float*)d_in[1];
  const float* qw    = (const float*)d_in[2];
  const float* qb    = (const float*)d_in[3];
  const float* kw    = (const float*)d_in[4];
  const float* kb    = (const float*)d_in[5];
  const float* vw    = (const float*)d_in[6];
  const float* vb    = (const float*)d_in[7];
  const float* fw    = (const float*)d_in[8];
  const float* fb    = (const float*)d_in[9];
  float* out = (float*)d_out;

  char* ws = (char*)d_ws;
  size_t off = 0;
  auto alloc = [&](size_t bytes) -> char* {
    char* p = ws + off;
    off = (off + bytes + 255) & ~(size_t)255;
    return p;
  };
  float* S    = (float*)alloc((size_t)B_ * C_ * W_ * 4);
  float* R0   = (float*)alloc((size_t)B_ * C_ * W_ * 4);
  float* RH   = (float*)alloc((size_t)B_ * C_ * W_ * 4);
  float* kbuf = (float*)alloc((size_t)B_ * C8_ * WP * 4);
  float* vsum = (float*)alloc((size_t)B_ * C_ * WP * 4);
  float* mbuf = (float*)alloc((size_t)B_ * C_ * WP * 4);
  float* kqb  = (float*)alloc((size_t)B_ * WP * 4);
  float* ebuf = (float*)alloc((size_t)B_ * HP * WP * 4);
  bf16*  Bp   = (bf16*)alloc((size_t)KSTEPS * NTILES * 32 * 16 * 2);
  bf16*  cat  = (bf16*)alloc((size_t)B_ * H2 * W2 * CIN * 2);
  (void)ws_size; (void)in_sizes; (void)n_in; (void)out_size;

  reduce_front_k<<<B_ * C_, 128, 0, stream>>>(front, S, R0, RH);
  compute_k_k<<<B_ * WP, 256, 0, stream>>>(S, kw, kb, kbuf);
  compute_vsum_k<<<B_ * 16, 256, 0, stream>>>(S, R0, RH, vw, vb, vsum);
  compute_m_k<<<B_ * WP, 256, 0, stream>>>(kbuf, qw, qb, mbuf, kqb);
  compute_e_k<<<B_ * HP, 256, 0, stream>>>(bev, mbuf, kqb, ebuf);
  normalize_e_k<<<B_ * WP, HP, 0, stream>>>(ebuf);
  {
    int tot = KSTEPS * NTILES * 32 * 16;
    pack_weights_k<<<(tot + 255) / 256, 256, 0, stream>>>(fw, Bp);
  }
  zero_halo_k<<<B_ * 516, 128, 0, stream>>>(cat);
  build_cat_k<<<B_ * H_, 256, 0, stream>>>(bev, ebuf, vsum, cat);
  copy16_k<<<(B_ * 65536 + 255) / 256, 256, 0, stream>>>((const float4*)bev, (float4*)out);
  conv_main_k<<<B_ * H_ * 2, 256, 0, stream>>>(cat, Bp, fb, out);
}
